// MultiChoiceRNN_5428838662591
// MI455X (gfx1250) — matrix operand order repaired and checked by reading, not compiled, not run on hardware
//
#include <hip/hip_runtime.h>
#include <hip/hip_bf16.h>

// ---------------- static problem sizes (match reference) ----------------
#define TT 64
#define BB 64
#define CC 2
#define KI 64       // input units I
#define HH 256
#define OO 3
#define GG 3
#define EH 204      // excitatory hidden units (sign +1 below, -1 at/above)
#define EXIST 205   // H - ZC : readout/attn columns that exist

typedef __attribute__((ext_vector_type(16))) _Float16 v16h;
typedef __attribute__((ext_vector_type(8)))  float    v8f;

__device__ __forceinline__ float spow(float x) {           // x**0.4, 0 for x<=0
  return (x > 0.f) ? __expf(0.4f * __logf(x)) : 0.f;
}
__device__ __forceinline__ float softplus(float x) {
  return (x > 20.f) ? x : __logf(1.f + __expf(x));
}
__device__ __forceinline__ int grp(int i) { return (i < 24) ? 0 : ((i < 48) ? 1 : 2); }
__device__ __forceinline__ float clip01(float x) { return fminf(fmaxf(x, 0.f), 1.f); }

// B operand (32x16 f16): activation chunk in column N=0, zeros elsewhere.
// Loads are UNCONDITIONAL b128 from LDS; the lane-select is applied to the value
// (one v_cndmask on `scale`), so no EXEC manipulation near the WMMA stream.
__device__ __forceinline__ v16h make_bvec(const float* __restrict__ vec, int k0, int lane) {
  const int khalf = lane >> 4;
  const float scale = ((lane & 15) == 0) ? 1.f : 0.f;
  const float4* vp = (const float4*)(vec + k0 + khalf * 8);
  v16h bv;
#pragma unroll
  for (int q = 0; q < 4; ++q) {
    const float4 f = vp[q + (q & 2)];
    bv[4 * q + 0] = (_Float16)(f.x * scale);
    bv[4 * q + 1] = (_Float16)(f.y * scale);
    bv[4 * q + 2] = (_Float16)(f.z * scale);
    bv[4 * q + 3] = (_Float16)(f.w * scale);
  }
  return bv;
}

// One 16-row x 32-K matvec chunk through the matrix pipe.
// A layout (16-bit 16x32): lanes 0-15 hold K {0..7,16..23}, lanes 16-31 hold K {8..15,24..31}.
// A-operand streamed as 4x global_load_b128 per call.
__device__ __forceinline__ v8f mv_chunk(const float* __restrict__ wrow, int ldw, int k0,
                                        v16h bv, v8f acc, int lane) {
  const int m = lane & 15;
  const int khalf = lane >> 4;
  const float4* wp = (const float4*)(wrow + (size_t)m * ldw + k0 + khalf * 8);
  const float4 f0 = wp[0];
  const float4 f1 = wp[1];
  const float4 f2 = wp[4];   // +16 floats
  const float4 f3 = wp[5];
  v16h a;
  a[0]  = (_Float16)f0.x; a[1]  = (_Float16)f0.y; a[2]  = (_Float16)f0.z; a[3]  = (_Float16)f0.w;
  a[4]  = (_Float16)f1.x; a[5]  = (_Float16)f1.y; a[6]  = (_Float16)f1.z; a[7]  = (_Float16)f1.w;
  a[8]  = (_Float16)f2.x; a[9]  = (_Float16)f2.y; a[10] = (_Float16)f2.z; a[11] = (_Float16)f2.w;
  a[12] = (_Float16)f3.x; a[13] = (_Float16)f3.y; a[14] = (_Float16)f3.z; a[15] = (_Float16)f3.w;
  return __builtin_amdgcn_wmma_f32_16x16x32_f16(false, a, false, bv, (short)0, acc,
                                                false, false);
}

// wave32 butterfly sum
__device__ __forceinline__ float wave_sum(float v) {
#pragma unroll
  for (int m = 16; m > 0; m >>= 1) v += __shfl_xor(v, m, 32);
  return v;
}

// Rebuild per-sample plastic weight state in workspace every call (deterministic).
__global__ void init_ws_kernel(const float* __restrict__ w_x, const float* __restrict__ w_h,
                               float* __restrict__ wx_all, float* __restrict__ wh_all) {
  const size_t nwx = (size_t)BB * CC * HH * KI;  // 2,097,152
  const size_t nwh = (size_t)BB * HH * HH;       // 4,194,304
  const size_t stride = (size_t)gridDim.x * blockDim.x;
  for (size_t e = (size_t)blockIdx.x * blockDim.x + threadIdx.x; e < nwx; e += stride)
    wx_all[e] = fmaxf(w_x[e & (CC * HH * KI - 1)], 0.f);
  for (size_t e = (size_t)blockIdx.x * blockDim.x + threadIdx.x; e < nwh; e += stride)
    wh_all[e] = fmaxf(w_h[e & (HH * HH - 1)], 0.f);
}

__global__ __launch_bounds__(256)
void rnn_kernel(const float* __restrict__ x, const float* __restrict__ R,
                const float* __restrict__ w_x, const float* __restrict__ w_h,
                const float* __restrict__ b_h, const float* __restrict__ w_attn,
                const float* __restrict__ b_attn, const float* __restrict__ w_o,
                const float* __restrict__ b_o, const float* __restrict__ w_v,
                const float* __restrict__ b_v, const float* __restrict__ kin,
                const float* __restrict__ krec, const float* __restrict__ kfb,
                float* __restrict__ out, float* __restrict__ wx_all,
                float* __restrict__ wh_all) {
  const int b = blockIdx.x;               // one workgroup per sample
  const int tid = threadIdx.x;            // 256 threads = 8 wave32s; tid == hidden index h
  const int lane = tid & 31;
  const int wave = tid >> 5;
  float* wx_s = wx_all + (size_t)b * (CC * HH * KI);
  float* wh_s = wh_all + (size_t)b * (HH * HH);

  __shared__ float s_out[HH], s_sout[HH], s_state[HH], s_pre[HH], s_post[HH], s_acc[HH];
  __shared__ float s_modx[CC * KI], s_pmodx[CC * KI];
  __shared__ float s_wa[GG * HH];             // per-sample plastic attention weights (LDS-resident)
  __shared__ float s_kin[CC * HH * GG], s_kfb[GG * HH];
  __shared__ float s_attn[GG], s_pattn[GG], s_logit[GG], s_oacc[4];

  s_out[tid] = 0.f;
  s_state[tid] = 0.f;
  for (int e = tid; e < GG * HH; e += 256) {
    s_wa[e] = fmaxf(w_attn[e], 0.f);
    s_kfb[e] = kfb[e];
  }
  for (int e = tid; e < CC * HH * GG; e += 256) s_kin[e] = kin[e];
  __syncthreads();

  for (int t = 0; t < TT; ++t) {
    const float rr = 0.02f * R[t * BB + b];   // DT * R_t (uniform, L2 hit)
    if (tid < GG) s_logit[tid] = 0.f;
    if (tid < 4) s_oacc[tid] = 0.f;
    __syncthreads();

    // ---- attention logits over previous output; fold EI sign into sout ----
    {
      const float o = s_out[tid];
      const float sgn = (tid < EH) ? 1.f : -1.f;
      s_sout[tid] = sgn * o;
      const float so = (tid < EXIST) ? sgn * o : 0.f;
      float p0 = wave_sum(s_wa[0 * HH + tid] * so);
      float p1 = wave_sum(s_wa[1 * HH + tid] * so);
      float p2 = wave_sum(s_wa[2 * HH + tid] * so);
      if (lane == 0) {
        atomicAdd(&s_logit[0], p0);
        atomicAdd(&s_logit[1], p1);
        atomicAdd(&s_logit[2], p2);
      }
    }
    __syncthreads();
    if (tid == 0) {
      float l0 = s_logit[0] + b_attn[0];
      float l1 = s_logit[1] + b_attn[1];
      float l2 = s_logit[2] + b_attn[2];
      float mx = fmaxf(l0, fmaxf(l1, l2));
      float e0 = __expf(l0 - mx), e1 = __expf(l1 - mx), e2 = __expf(l2 - mx);
      float inv = 1.f / (e0 + e1 + e2);
      s_attn[0] = e0 * inv; s_attn[1] = e1 * inv; s_attn[2] = e2 * inv;
      s_pattn[0] = spow(s_attn[0]); s_pattn[1] = spow(s_attn[1]); s_pattn[2] = spow(s_attn[2]);
    }
    __syncthreads();
    if (tid < CC * KI) {                      // modulated input (C*I = 128 values)
      const int c = tid >> 6, i = tid & 63;
      const float m = x[(((size_t)t * BB + b) * CC + c) * KI + i] * s_attn[grp(i)];
      s_modx[tid] = m;
      s_pmodx[tid] = spow(m);
    }
    __syncthreads();

    // ---- ext + rec matvec through WMMA: each wave owns 32 rows (2 tiles) ----
    {
      const int r0 = wave * 32;
      v8f acc0 = {};
      v8f acc1 = {};
      for (int k0 = 0; k0 < HH; k0 += 32) {   // recurrent: K over signed output
        const v16h bv = make_bvec(s_sout, k0, lane);
        acc0 = mv_chunk(wh_s + (size_t)r0 * HH, HH, k0, bv, acc0, lane);
        acc1 = mv_chunk(wh_s + (size_t)(r0 + 16) * HH, HH, k0, bv, acc1, lane);
      }
#pragma unroll
      for (int c = 0; c < CC; ++c) {          // external: K over modulated input
#pragma unroll
        for (int k0 = 0; k0 < KI; k0 += 32) {
          const v16h bv = make_bvec(s_modx + c * KI, k0, lane);
          const float* base = wx_s + (size_t)c * HH * KI;
          acc0 = mv_chunk(base + (size_t)r0 * KI, KI, k0, bv, acc0, lane);
          acc1 = mv_chunk(base + (size_t)(r0 + 16) * KI, KI, k0, bv, acc1, lane);
        }
      }
      if ((lane & 15) == 0) {                 // column N=0 lives in lanes 0 (M=r) and 16 (M=8+r)
        const int off = (lane >> 4) * 8;
#pragma unroll
        for (int r = 0; r < 8; ++r) {
          s_acc[r0 + off + r] = acc0[r];
          s_acc[r0 + 16 + off + r] = acc1[r];
        }
      }
    }
    __syncthreads();

    // ---- state update (thread == hidden unit); remove the masked diagonal ----
    float no;
    {
      const float diag = wh_s[(size_t)tid * HH + tid] * s_sout[tid];
      const float tot = s_acc[tid] - diag + b_h[tid];
      const float ns = 0.8f * s_state[tid] + 0.2f * tot;
      no = softplus(ns);
      s_state[tid] = ns;
      s_pre[tid] = spow(s_out[tid]);          // pow of previous output
      s_post[tid] = spow(no);
      s_out[tid] = no;                        // own index only: race-free
    }
    __syncthreads();

    // ---- readout reductions: wave32 butterfly + one LDS atomic per wave ----
    {
      const float noe = (tid < EXIST) ? no : 0.f;
      float q0 = wave_sum(noe * fmaxf(w_o[0 * HH + tid], 0.f));
      float q1 = wave_sum(noe * fmaxf(w_o[1 * HH + tid], 0.f));
      float q2 = wave_sum(noe * fmaxf(w_o[2 * HH + tid], 0.f));
      float q3 = wave_sum(noe * fmaxf(w_v[tid], 0.f));
      if (lane == 0) {
        atomicAdd(&s_oacc[0], q0);
        atomicAdd(&s_oacc[1], q1);
        atomicAdd(&s_oacc[2], q2);
        atomicAdd(&s_oacc[3], q3);
      }
    }

    // ---- plasticity: b128 streaming RMW over L2-resident per-sample weights ----
    for (int e = tid * 4; e < HH * HH; e += 1024) {
      __builtin_prefetch(&wh_s[e + 4096], 0, 1);
      float4 w  = *(const float4*)&wh_s[e];
      float4 wb = *(const float4*)&w_h[e];
      float4 kr = *(const float4*)&krec[e];
      const int h = e >> 8, j = e & 255;
      const float ph = rr * s_post[h];
      const float4 pr = *(const float4*)&s_pre[j];
      w.x = clip01(0.02f * fmaxf(wb.x, 0.f) + 0.98f * w.x + ph * kr.x * spow(w.x) * pr.x);
      w.y = clip01(0.02f * fmaxf(wb.y, 0.f) + 0.98f * w.y + ph * kr.y * spow(w.y) * pr.y);
      w.z = clip01(0.02f * fmaxf(wb.z, 0.f) + 0.98f * w.z + ph * kr.z * spow(w.z) * pr.z);
      w.w = clip01(0.02f * fmaxf(wb.w, 0.f) + 0.98f * w.w + ph * kr.w * spow(w.w) * pr.w);
      *(float4*)&wh_s[e] = w;
    }
    for (int e = tid * 4; e < CC * HH * KI; e += 1024) {
      float4 w  = *(const float4*)&wx_s[e];
      float4 wb = *(const float4*)&w_x[e];
      const int c = e >> 14, rem = e & 16383, h = rem >> 6, i = rem & 63;
      const float ph = rr * s_post[h];
      const float4 pm = *(const float4*)&s_pmodx[c * KI + i];
      const float* kp = &s_kin[(c * HH + h) * GG];
      w.x = clip01(0.02f * fmaxf(wb.x, 0.f) + 0.98f * w.x + ph * kp[grp(i + 0)] * spow(w.x) * pm.x);
      w.y = clip01(0.02f * fmaxf(wb.y, 0.f) + 0.98f * w.y + ph * kp[grp(i + 1)] * spow(w.y) * pm.y);
      w.z = clip01(0.02f * fmaxf(wb.z, 0.f) + 0.98f * w.z + ph * kp[grp(i + 2)] * spow(w.z) * pm.z);
      w.w = clip01(0.02f * fmaxf(wb.w, 0.f) + 0.98f * w.w + ph * kp[grp(i + 3)] * spow(w.w) * pm.w);
      *(float4*)&wx_s[e] = w;
    }
    for (int e = tid; e < GG * HH; e += 256) {
      const float w = s_wa[e];
      const int g = e >> 8, h = e & 255;
      const float w0 = fmaxf(w_attn[e], 0.f);
      const float dw = rr * s_kfb[e] * spow(w) * s_pattn[g] * s_pre[h];
      s_wa[e] = clip01(0.02f * w0 + 0.98f * w + dw);
    }
    __syncthreads();

    if (tid < 4) {                            // ys[t,b,:] = [o0,o1,o2,v]
      const float bias = (tid < OO) ? b_o[tid] : b_v[0];
      out[(((size_t)t * BB) + b) * 4 + tid] = s_oacc[tid] + bias;
    }
    __syncthreads();
  }
}

extern "C" void kernel_launch(void* const* d_in, const int* in_sizes, int n_in,
                              void* d_out, int out_size, void* d_ws, size_t ws_size,
                              hipStream_t stream) {
  (void)in_sizes; (void)n_in; (void)out_size; (void)ws_size;
  const float* x      = (const float*)d_in[0];
  const float* R      = (const float*)d_in[1];
  const float* w_x    = (const float*)d_in[2];
  const float* w_h    = (const float*)d_in[3];
  const float* b_h    = (const float*)d_in[4];
  const float* w_attn = (const float*)d_in[5];
  const float* b_attn = (const float*)d_in[6];
  const float* w_o    = (const float*)d_in[7];
  const float* b_o    = (const float*)d_in[8];
  const float* w_v    = (const float*)d_in[9];
  const float* b_v    = (const float*)d_in[10];
  const float* kin    = (const float*)d_in[11];
  const float* krec   = (const float*)d_in[12];
  const float* kfb    = (const float*)d_in[13];
  float* out = (float*)d_out;

  float* wx_all = (float*)d_ws;                             // B*C*H*I floats
  float* wh_all = wx_all + (size_t)BB * CC * HH * KI;       // B*H*H floats

  init_ws_kernel<<<2048, 256, 0, stream>>>(w_x, w_h, wx_all, wh_all);
  rnn_kernel<<<BB, 256, 0, stream>>>(x, R, w_x, w_h, b_h, w_attn, b_attn, w_o, b_o,
                                     w_v, b_v, kin, krec, kfb, out, wx_all, wh_all);
}
